// FGSBIR_GAT_26912265077428
// MI455X (gfx1250) — matrix operand order repaired and checked by reading, not compiled, not run on hardware
//
#include <hip/hip_runtime.h>
#include <hip/hip_bf16.h>

#define NN 256
#define HH 8
#define FF 256
#define DG 2048
#define BB 64
#define DVIS 2048
#define DLBL 512
#define DEMB 512
#define LRELU_A 0.2f

typedef __attribute__((ext_vector_type(16))) __bf16 v16bf;
typedef __attribute__((ext_vector_type(8)))  float  v8f;

static __device__ __forceinline__ unsigned short f32_to_bf16(float x) {
    unsigned int u = __float_as_uint(x);
    unsigned int r = (u + 0x7FFFu + ((u >> 16) & 1u)) >> 16;   // round-to-nearest-even
    return (unsigned short)r;
}

// ---------------------------------------------------------------------------
// Kernel A: Wh_vis[b, h*F+f] = visual[b,:] @ W_gat[512:, h, f];  + srcV/dstV
// One block per b; thread t owns f=t for all 8 heads.
// ---------------------------------------------------------------------------
__global__ __launch_bounds__(256) void proj_vis_kernel(
    const float* __restrict__ visual, const float* __restrict__ Wg,
    const float* __restrict__ a_src, const float* __restrict__ a_dst,
    float* __restrict__ whVis, float* __restrict__ srcV, float* __restrict__ dstV)
{
    const int b = blockIdx.x;
    const int t = threadIdx.x;
    __shared__ float sRow[DVIS];
    __shared__ float sRed[HH][256];

    #pragma unroll
    for (int q = 0; q < 8; ++q) sRow[t + 256*q] = visual[b*DVIS + t + 256*q];
    __syncthreads();

    float acc[8] = {0,0,0,0,0,0,0,0};
    const float* Wcol = Wg + (long)DLBL * DG + t;   // rows d=512.. of W_gat
    for (int d = 0; d < DVIS; ++d) {
        const float x = sRow[d];
        const float* wr = Wcol + (long)d * DG;
        #pragma unroll
        for (int q = 0; q < 8; ++q) acc[q] = fmaf(x, wr[256*q], acc[q]);
    }
    #pragma unroll
    for (int q = 0; q < 8; ++q) whVis[b*DG + q*FF + t] = acc[q];

    // srcV reduction (deterministic tree)
    #pragma unroll
    for (int q = 0; q < 8; ++q) sRed[q][t] = acc[q] * a_src[q*FF + t];
    __syncthreads();
    for (int s = 128; s > 0; s >>= 1) {
        if (t < s) {
            #pragma unroll
            for (int q = 0; q < 8; ++q) sRed[q][t] += sRed[q][t + s];
        }
        __syncthreads();
    }
    if (t < 8) srcV[b*HH + t] = sRed[t][0];
    __syncthreads();

    #pragma unroll
    for (int q = 0; q < 8; ++q) sRed[q][t] = acc[q] * a_dst[q*FF + t];
    __syncthreads();
    for (int s = 128; s > 0; s >>= 1) {
        if (t < s) {
            #pragma unroll
            for (int q = 0; q < 8; ++q) sRed[q][t] += sRed[q][t + s];
        }
        __syncthreads();
    }
    if (t < 8) dstV[b*HH + t] = sRed[t][0];
}

// ---------------------------------------------------------------------------
// Kernel B: Wh_lbl[n, h, f] = labels[n,:] @ W_gat[0:512, h, f]
// Written as bf16 pre-swizzled WMMA B-fragments:
//   vfrag[(((h*8 + n/32)*16 + f/16)*32 + lane)*16 + half]
//   lane = (f&15) + 16*((n&31)>>4), half = (n&31)&15
// Also srcL[h*N+n], dstL[h*N+n].
// ---------------------------------------------------------------------------
__global__ __launch_bounds__(256) void proj_lbl_kernel(
    const float* __restrict__ labels, const float* __restrict__ Wg,
    const float* __restrict__ a_src, const float* __restrict__ a_dst,
    unsigned short* __restrict__ vfrag, float* __restrict__ srcL, float* __restrict__ dstL)
{
    const int n = blockIdx.x;
    const int t = threadIdx.x;
    __shared__ float sRow[DLBL];
    __shared__ float sRed[HH][256];

    sRow[t] = labels[n*DLBL + t];
    sRow[t + 256] = labels[n*DLBL + 256 + t];
    __syncthreads();

    float acc[8] = {0,0,0,0,0,0,0,0};
    const float* Wcol = Wg + t;
    for (int d = 0; d < DLBL; ++d) {
        const float x = sRow[d];
        const float* wr = Wcol + (long)d * DG;
        #pragma unroll
        for (int q = 0; q < 8; ++q) acc[q] = fmaf(x, wr[256*q], acc[q]);
    }

    // bf16 B-fragment store (j = n is the K dimension of the attention matmul)
    const int jj   = n & 31;
    const int kt   = n >> 5;
    const int lane = (t & 15) + 16 * (jj >> 4);
    const int half = jj & 15;
    const int tcf  = t >> 4;
    #pragma unroll
    for (int q = 0; q < 8; ++q) {
        const long idx = ((((long)(q*8 + kt)*16 + tcf)*32 + lane)*16 + half);
        vfrag[idx] = f32_to_bf16(acc[q]);
    }

    #pragma unroll
    for (int q = 0; q < 8; ++q) sRed[q][t] = acc[q] * a_src[q*FF + t];
    __syncthreads();
    for (int s = 128; s > 0; s >>= 1) {
        if (t < s) {
            #pragma unroll
            for (int q = 0; q < 8; ++q) sRed[q][t] += sRed[q][t + s];
        }
        __syncthreads();
    }
    if (t < 8) srcL[t*NN + n] = sRed[t][0];
    __syncthreads();

    #pragma unroll
    for (int q = 0; q < 8; ++q) sRed[q][t] = acc[q] * a_dst[q*FF + t];
    __syncthreads();
    for (int s = 128; s > 0; s >>= 1) {
        if (t < s) {
            #pragma unroll
            for (int q = 0; q < 8; ++q) sRed[q][t] += sRed[q][t + s];
        }
        __syncthreads();
    }
    if (t < 8) dstL[t*NN + n] = sRed[t][0];
}

// ---------------------------------------------------------------------------
// Kernel C: fused masked-softmax attention + P@V via WMMA + bias + ELU.
// Block = (b, h, 32-row panel). 8 waves, each computes 4 16x16 output tiles.
// out[b, i, h*F + f] = ELU( sum_j P[i,j] * Wh_lbl[j,h,f] + Wh_vis[b,h,f] )
// ---------------------------------------------------------------------------
__global__ __launch_bounds__(256) void gat_attn_kernel(
    const float* __restrict__ adj,
    const float* __restrict__ srcL, const float* __restrict__ dstL,
    const float* __restrict__ srcV, const float* __restrict__ dstV,
    const float* __restrict__ whVis,
    const unsigned short* __restrict__ vfrag,
    float* __restrict__ outBuf)
{
    const int blk = blockIdx.x;
    const int b   = blk >> 6;
    const int h   = (blk >> 3) & 7;
    const int pnl = blk & 7;
    const int i0  = pnl * 32;
    const int t   = threadIdx.x;

    __shared__ float sDst[NN];
    __shared__ float sSrc[32];
    __shared__ float sM[32];
    __shared__ float sInvS[32];
    __shared__ float sRed[256];
    __shared__ __align__(16) unsigned short sP[16*32*16];   // 16 A-fragments, 16 KB

    const float svb = srcV[b*HH + h];
    const float dvb = dstV[b*HH + h];
    sDst[t] = dstL[h*NN + t] + dvb;
    if (t < 32) sSrc[t] = srcL[h*NN + i0 + t] + svb;
    __syncthreads();

    // --- softmax stats: 8 threads per row (r = t>>3, part p = t&7) ---
    const int r = t >> 3, p = t & 7;
    const float* adjRow = adj + (i0 + r) * NN;
    const float srcR = sSrc[r];

    float m = -3.0e38f;
    for (int j = p*32; j < p*32 + 32; ++j) {
        if (adjRow[j] > 0.f) {
            float e = srcR + sDst[j];
            e = e > 0.f ? e : LRELU_A * e;
            m = fmaxf(m, e);
        }
    }
    sRed[t] = m;
    __syncthreads();
    if (t < 32) {
        float mm = sRed[t*8];
        #pragma unroll
        for (int q = 1; q < 8; ++q) mm = fmaxf(mm, sRed[t*8 + q]);
        sM[t] = mm;
    }
    __syncthreads();

    const float mr = sM[r];
    float s = 0.f;
    for (int j = p*32; j < p*32 + 32; ++j) {
        if (adjRow[j] > 0.f) {
            float e = srcR + sDst[j];
            e = e > 0.f ? e : LRELU_A * e;
            s += __expf(e - mr);
        }
    }
    sRed[t] = s;
    __syncthreads();
    if (t < 32) {
        float ss = 0.f;
        #pragma unroll
        for (int q = 0; q < 8; ++q) ss += sRed[t*8 + q];
        sInvS[t] = 1.0f / ss;
    }
    __syncthreads();

    // --- build bf16 A-fragments of P in LDS (contiguous k = 16*(lane>>4) + half order) ---
    #pragma unroll 4
    for (int q = 0; q < 32; ++q) {
        const int idx  = t*32 + q;
        const int half = idx & 15;
        const int lane = (idx >> 4) & 31;
        const int frag = idx >> 9;             // 0..15 = tr*8 + kt
        const int tr = frag >> 3, kt = frag & 7;
        const int mrow = lane & 15, g = lane >> 4;
        const int k = half + (g << 4);
        const int row = tr*16 + mrow;
        const int j = kt*32 + k;
        unsigned short v = 0;
        if (adj[(i0 + row)*NN + j] > 0.f) {
            float e = sSrc[row] + sDst[j];
            e = e > 0.f ? e : LRELU_A * e;
            v = f32_to_bf16(__expf(e - sM[row]) * sInvS[row]);
        }
        sP[idx] = v;
    }
    __syncthreads();

    // --- WMMA: 8 waves x 4 tiles; K loop over 8 bf16 16x16x32 steps ---
    const int w = t >> 5, lane = t & 31;
    const uint4* sPv = (const uint4*)sP;
    const uint4* gVf = (const uint4*)vfrag;

    for (int q = 0; q < 4; ++q) {
        const int tile = w*4 + q;
        const int tr = tile >> 4, tc = tile & 15;
        v8f acc = {0.f,0.f,0.f,0.f,0.f,0.f,0.f,0.f};
        #pragma unroll
        for (int kt = 0; kt < 8; ++kt) {
            union { uint4 u[2]; v16bf v; } A, Bm;
            const int af = ((tr*8 + kt)*32 + lane) * 2;                    // uint4 units
            A.u[0] = sPv[af];   A.u[1] = sPv[af + 1];
            const long bf = (((long)(h*8 + kt)*16 + tc)*32 + lane) * 2;    // uint4 units
            Bm.u[0] = gVf[bf];  Bm.u[1] = gVf[bf + 1];
            acc = __builtin_amdgcn_wmma_f32_16x16x32_bf16(
                      false, A.v, false, Bm.v, (short)0, acc, false, false);
        }
        // epilogue: C layout -> f = tc*16 + (lane&15); i = i0+tr*16 + rr + 8*(lane>>4)
        const int fg = h*FF + tc*16 + (lane & 15);
        const float bias = whVis[b*DG + fg];
        float* outP = outBuf + ((long)(b*NN + i0 + tr*16) * DG) + fg;
        const int mbase = (lane >> 4) * 8;
        #pragma unroll
        for (int rr = 0; rr < 8; ++rr) {
            float o = acc[rr] + bias;
            o = o > 0.f ? o : (__expf(o) - 1.0f);        // ELU, alpha=1
            outP[(long)(mbase + rr) * DG] = o;
        }
    }
}

// ---------------------------------------------------------------------------
// Kernel D: attention pooling over nodes + final FC. One block per b.
// ---------------------------------------------------------------------------
__global__ __launch_bounds__(256) void pool_fc_kernel(
    const float* __restrict__ outBuf, const float* __restrict__ poolq,
    const float* __restrict__ fcW, const float* __restrict__ fcb,
    float* __restrict__ y)
{
    const int b = blockIdx.x;
    const int t = threadIdx.x;
    __shared__ float sQ[DG];
    __shared__ float sT[NN];
    __shared__ float sPw[NN];
    __shared__ float sPool[DG];
    __shared__ float sR[256];

    #pragma unroll
    for (int q = 0; q < 8; ++q) sQ[t + 256*q] = poolq[t + 256*q];
    __syncthreads();

    {   // t[n] = out[b,n,:] . pool_q   (thread t owns row n = t)
        const float* row = outBuf + (long)(b*NN + t) * DG;
        float acc = 0.f;
        for (int d = 0; d < DG; ++d) acc = fmaf(row[d], sQ[d], acc);
        sT[t] = acc;
    }
    __syncthreads();

    sR[t] = sT[t];
    __syncthreads();
    for (int s = 128; s > 0; s >>= 1) { if (t < s) sR[t] = fmaxf(sR[t], sR[t+s]); __syncthreads(); }
    const float mx = sR[0];
    __syncthreads();
    const float ex = __expf(sT[t] - mx);
    sR[t] = ex;
    __syncthreads();
    for (int s = 128; s > 0; s >>= 1) { if (t < s) sR[t] += sR[t+s]; __syncthreads(); }
    const float inv = 1.0f / sR[0];
    sPw[t] = ex * inv;
    __syncthreads();

    // pooled[d] = sum_n pw[n] * out[b,n,d]   (coalesced over d)
    #pragma unroll
    for (int q = 0; q < 8; ++q) {
        const int d = t + 256*q;
        const float* col = outBuf + (long)b*NN*DG + d;
        float acc = 0.f;
        for (int n2 = 0; n2 < NN; ++n2) acc = fmaf(sPw[n2], col[(long)n2 * DG], acc);
        sPool[d] = acc;
    }
    __syncthreads();

    // y[b,e] = pooled . fc_W[:,e] + fc_b[e]
    #pragma unroll
    for (int q = 0; q < 2; ++q) {
        const int e = t + 256*q;
        float acc = fcb[e];
        for (int d = 0; d < DG; ++d) acc = fmaf(sPool[d], fcW[(long)d*DEMB + e], acc);
        y[b*DEMB + e] = acc;
    }
}

// ---------------------------------------------------------------------------
extern "C" void kernel_launch(void* const* d_in, const int* in_sizes, int n_in,
                              void* d_out, int out_size, void* d_ws, size_t ws_size,
                              hipStream_t stream) {
    const float* visual = (const float*)d_in[0];   // (64, 2048)
    const float* labels = (const float*)d_in[1];   // (256, 512)
    const float* adj    = (const float*)d_in[2];   // (256, 256)
    const float* Wg     = (const float*)d_in[3];   // (2560, 8, 256)
    const float* a_src  = (const float*)d_in[4];   // (8, 256)
    const float* a_dst  = (const float*)d_in[5];   // (8, 256)
    const float* poolq  = (const float*)d_in[6];   // (2048,)
    const float* fcW    = (const float*)d_in[7];   // (2048, 512)
    const float* fcb    = (const float*)d_in[8];   // (512,)
    float* y = (float*)d_out;                      // (64, 512)

    char* ws = (char*)d_ws;
    float* whVis = (float*)(ws + 0);                       // 512 KB
    float* srcV  = (float*)(ws + 0x80000);                 //   2 KB
    float* dstV  = (float*)(ws + 0x80000 + 2048);          //   2 KB
    float* srcL  = (float*)(ws + 0x80000 + 4096);          //   8 KB
    float* dstL  = (float*)(ws + 0x80000 + 4096 + 8192);   //   8 KB
    unsigned short* vfrag = (unsigned short*)(ws + 0xA0000);  // 1 MB bf16 B-fragments
    float* outBuf = (float*)(ws + 0x200000);               // 134 MB intermediate

    proj_vis_kernel<<<dim3(BB),       dim3(256), 0, stream>>>(visual, Wg, a_src, a_dst, whVis, srcV, dstV);
    proj_lbl_kernel<<<dim3(NN),       dim3(256), 0, stream>>>(labels, Wg, a_src, a_dst, vfrag, srcL, dstL);
    gat_attn_kernel<<<dim3(BB*HH*8),  dim3(256), 0, stream>>>(adj, srcL, dstL, srcV, dstV, whVis, vfrag, outBuf);
    pool_fc_kernel<<<dim3(BB),        dim3(256), 0, stream>>>(outBuf, poolq, fcW, fcb, y);
}
